// WavenetModel_47450798686596
// MI455X (gfx1250) — compile-verified, matrix operand order checked by reading, not run on hardware
//
#include <hip/hip_runtime.h>
#include <hip/hip_bf16.h>
#include <math.h>

#define LAYERS 10
#define NL 30
#define RC 32
#define SC 256
#define CLASSES 256
#define BB 4
#define TT 32768

typedef unsigned int u32;
typedef __attribute__((ext_vector_type(16))) __bf16 v16bf;
typedef __attribute__((ext_vector_type(2)))  __bf16 v2bf;
typedef __attribute__((ext_vector_type(8)))  u32    v8u;
typedef __attribute__((ext_vector_type(8)))  float  v8f;

union ABFrag { v8u u; v16bf b; };

#ifdef __has_builtin
#if __has_builtin(__builtin_amdgcn_cvt_pk_bf16_f32)
#define HAVE_CVT_PK_BF16 1
#endif
#if __has_builtin(__builtin_amdgcn_tanhf)
#define HAVE_TANHF 1
#endif
#endif

__device__ __forceinline__ u32 packbf2(float lo, float hi) {
#ifdef HAVE_CVT_PK_BF16
  auto t = __builtin_amdgcn_cvt_pk_bf16_f32(lo, hi);
  return __builtin_bit_cast(u32, t);
#else
  v2bf v = {(__bf16)lo, (__bf16)hi};   // build_vector(fptrunc x2) -> v_cvt_pk_bf16_f32
  return __builtin_bit_cast(u32, v);
#endif
}

__device__ __forceinline__ float fast_rcp(float x) {
  return __builtin_amdgcn_rcpf(x);
}

__device__ __forceinline__ float fast_tanh(float x) {
#ifdef HAVE_TANHF
  return __builtin_amdgcn_tanhf(x);
#else
  float e = __expf(2.0f * x);
  return 1.0f - 2.0f * fast_rcp(e + 1.0f);
#endif
}

__device__ __forceinline__ float fast_sigmoid(float x) {
  return fast_rcp(1.0f + __expf(-x));
}

// K-pair base for fragment reg r (0..7), half-wave group g (0/1) (ISA 7.12.2)
__device__ __forceinline__ int kpair(int r, int g) {
  return ((r < 4) ? (2 * r) : (2 * r + 8)) + (g << 3);
}

// A fragment from pre-swizzled weights: frag is 256 dwords, lane-major.
__device__ __forceinline__ v16bf load_Afrag_sw(const u32* __restrict__ frag) {
  const int lane = threadIdx.x & 31;
  const uint4* p = (const uint4*)(frag + lane * 8);
  uint4 lo = p[0], hi = p[1];
  ABFrag f;
  f.u[0] = lo.x; f.u[1] = lo.y; f.u[2] = lo.z; f.u[3] = lo.w;
  f.u[4] = hi.x; f.u[5] = hi.y; f.u[6] = hi.z; f.u[7] = hi.w;
  return f.b;
}

// B fragment from pair-packed LDS tile: base = uint tile[n=16][p=16].
// Lane (n,g) reads dwords p=4g..4g+3 and 8+4g..11+4g of row n (contiguous 16B).
__device__ __forceinline__ v16bf load_Bfrag_sw(const u32* base) {
  const int lane = threadIdx.x & 31;
  const int n = lane & 15;
  const int g = (lane >> 4) & 1;
  const u32* q = base + n * 16 + (g << 2);
  uint4 lo = *(const uint4*)q;
  uint4 hi = *(const uint4*)(q + 8);
  ABFrag f;
  f.u[0] = lo.x; f.u[1] = lo.y; f.u[2] = lo.z; f.u[3] = lo.w;
  f.u[4] = hi.x; f.u[5] = hi.y; f.u[6] = hi.z; f.u[7] = hi.w;
  return f.b;
}

__device__ __forceinline__ v8f wmma_bf16(v16bf a, v16bf b, v8f c) {
  return __builtin_amdgcn_wmma_f32_16x16x32_bf16(false, a, false, b, (short)0, c, false, false);
}

// Stage a f32 channel pair (rows row0,row1; 16 aligned f32 each) into LDS tile
// column slot: dst = &tile[0][pl], row stride 16 dwords.
__device__ __forceinline__ void stage_pair(u32* dst, const float* row0, const float* row1) {
  const float4* r0 = (const float4*)row0;
  const float4* r1 = (const float4*)row1;
#pragma unroll
  for (int q = 0; q < 4; ++q) {
    float4 a = r0[q], c = r1[q];
    dst[(q * 4 + 0) * 16] = packbf2(a.x, c.x);
    dst[(q * 4 + 1) * 16] = packbf2(a.y, c.y);
    dst[(q * 4 + 2) * 16] = packbf2(a.z, c.z);
    dst[(q * 4 + 3) * 16] = packbf2(a.w, c.w);
  }
}

// ---------------- weight prep: f32 -> bf16 swizzled fragments ---------------
__global__ void prep_kernel(const float* __restrict__ Wc_f, const float* __restrict__ Wf_f,
                            const float* __restrict__ Wr_f, const float* __restrict__ Ws_f,
                            const float* __restrict__ We1_f, const float* __restrict__ We2_f,
                            const float* __restrict__ bs_f,
                            u32* __restrict__ WcS, u32* __restrict__ WfS,
                            u32* __restrict__ WrS, u32* __restrict__ WsS,
                            u32* __restrict__ We1S, u32* __restrict__ We2S,
                            float* __restrict__ bsum) {
  const int j = blockIdx.x * 256 + threadIdx.x;
  const int e = j & 255, lane = e >> 3, r = e & 7;
  const int ml = lane & 15, g = lane >> 4;
  const int kp = kpair(r, g);
  switch (blockIdx.y) {
    case 0: {  // W_causal [32][256] -> 16 frags (mt*8+kt)
      if (j < 16 * 256) {
        int fi = j >> 8, mt = fi >> 3, kt = fi & 7;
        int m = mt * 16 + ml, k = kt * 32 + kp;
        WcS[j] = packbf2(Wc_f[m * 256 + k], Wc_f[m * 256 + k + 1]);
      }
    } break;
    case 1: {  // Wf [NL][32][32][2] -> per layer 4 frags (mt*2+kt), K = tap*32+c
      if (j < NL * 4 * 256) {
        int i = j >> 10, rem = j & 1023, fi = rem >> 8, mt = fi >> 1, kt = fi & 1;
        int m = mt * 16 + ml, kidx = kt * 32 + kp, tap = kidx >> 5, c = kidx & 31;
        float v0 = Wf_f[((i * 32 + m) * 32 + c) * 2 + tap];
        float v1 = Wf_f[((i * 32 + m) * 32 + c + 1) * 2 + tap];
        WfS[j] = packbf2(v0, v1);
      }
    } break;
    case 2: {  // Wr [NL][32][16] -> per layer 2 frags (mt), zero-padded K 16..31
      if (j < NL * 2 * 256) {
        int i = j >> 9, rem = j & 511, mt = rem >> 8;
        int m = mt * 16 + ml, k = kp;
        float v0 = (k < 16) ? Wr_f[(i * 32 + m) * 16 + k] : 0.f;
        float v1 = (k + 1 < 16) ? Wr_f[(i * 32 + m) * 16 + k + 1] : 0.f;
        WrS[j] = packbf2(v0, v1);
      }
    } break;
    case 3: {  // Ws [NL][256][16] -> Wsall 240 frags (mt*15+kt), K = layer*16+c
      if (j < 240 * 256) {
        int fi = j >> 8, mt = fi / 15, kt = fi % 15;
        int m = mt * 16 + ml, k = kt * 32 + kp, i = k >> 4, c = k & 15;
        float v0 = Ws_f[(i * 256 + m) * 16 + c];
        float v1 = Ws_f[(i * 256 + m) * 16 + c + 1];
        WsS[j] = packbf2(v0, v1);
      }
    } break;
    case 4: {  // W_e1 [256][256] -> 128 frags (mt*8+kt)
      if (j < 128 * 256) {
        int fi = j >> 8, mt = fi >> 3, kt = fi & 7;
        int m = mt * 16 + ml, k = kt * 32 + kp;
        We1S[j] = packbf2(We1_f[m * 256 + k], We1_f[m * 256 + k + 1]);
      }
    } break;
    case 5: {
      if (j < 128 * 256) {
        int fi = j >> 8, mt = fi >> 3, kt = fi & 7;
        int m = mt * 16 + ml, k = kt * 32 + kp;
        We2S[j] = packbf2(We2_f[m * 256 + k], We2_f[m * 256 + k + 1]);
      }
    } break;
    case 6: {  // bsum[m] = sum_i bs[i][m]
      if (j < 256) {
        float s = 0.f;
        for (int i = 0; i < NL; ++i) s += bs_f[i * 256 + j];
        bsum[j] = s;
      }
    } break;
  }
}

// ---------------- front: x0 = Wc[32x256] @ y + bc ---------------------------
// xf: f32 [b][32][T] (residual adds). xq: pair-minor packed bf16 [b][t][16].
__global__ void __launch_bounds__(64) front_kernel(const float* __restrict__ y,
                                                   const u32* __restrict__ WcS,
                                                   const float* __restrict__ bc,
                                                   float* __restrict__ xf,
                                                   u32* __restrict__ xq) {
  __shared__ __align__(16) u32 tile[2][8][16][16];
  const int wave = threadIdx.x >> 5;
  const int lane = threadIdx.x & 31;
  const int b = blockIdx.y;
  const int t0 = blockIdx.x * 32 + wave * 16;
  u32 (*bt)[16][16] = tile[wave];

  const float* ybase = y + ((size_t)b * CLASSES) * TT + t0;
#pragma unroll
  for (int pp = 0; pp < 4; ++pp) {
    int p = lane + pp * 32;                       // pair 0..127
    stage_pair(&bt[p >> 4][0][p & 15],
               ybase + (size_t)(2 * p) * TT,
               ybase + (size_t)(2 * p + 1) * TT);
  }
  const int g = (lane >> 4) & 1;
  const int n = lane & 15;
  float* dstf = xf + ((size_t)b * RC) * TT + t0;
  u32* dstq = xq + ((size_t)b * TT + t0 + n) * 16;
#pragma unroll
  for (int mt = 0; mt < 2; ++mt) {
    v8f acc;
#pragma unroll
    for (int r = 0; r < 8; ++r) acc[r] = bc[mt * 16 + r + 8 * g];
    for (int kt = 0; kt < 8; ++kt)
      acc = wmma_bf16(load_Afrag_sw(WcS + (mt * 8 + kt) * 256),
                      load_Bfrag_sw(&bt[kt][0][0]), acc);
#pragma unroll
    for (int r = 0; r < 8; ++r)
      dstf[(size_t)(mt * 16 + r + 8 * g) * TT + n] = acc[r];
    uint4 uq;
    uq.x = packbf2(acc[0], acc[1]);
    uq.y = packbf2(acc[2], acc[3]);
    uq.z = packbf2(acc[4], acc[5]);
    uq.w = packbf2(acc[6], acc[7]);
    *(uint4*)(dstq + mt * 8 + 4 * g) = uq;
  }
}

// ---------------- residual layer --------------------------------------------
__global__ void __launch_bounds__(128) layer_kernel(const float* __restrict__ xf_in,
                                                    const u32* __restrict__ xq_in,
                                                    float* __restrict__ xf_out,
                                                    u32* __restrict__ xq_out,
                                                    u32* __restrict__ G2,
                                                    const u32* __restrict__ WfSi,
                                                    const float* __restrict__ bfv,
                                                    const u32* __restrict__ WrSi,
                                                    const float* __restrict__ brv,
                                                    int layer, int d) {
  __shared__ __align__(16) u32 xt[4][2][16][16];
  __shared__ __align__(16) u32 gt[4][16][16];
  const int wave = threadIdx.x >> 5;
  const int lane = threadIdx.x & 31;
  const int b = blockIdx.y;
  const int t0 = blockIdx.x * 64 + wave * 16;
  const u32* xqb = xq_in + (size_t)b * TT * 16;
  const int g = (lane >> 4) & 1;
  const int n = lane & 15;

  // stage concat (pure b128 copies): ktile0 = pairs at t-d (0 if t<0),
  // ktile1 = pairs at t.  Lane (n,g) covers quarters q = 2g and 2g+1.
#pragma unroll
  for (int s = 0; s < 2; ++s) {
    int q4 = (2 * g + s) * 4;                       // dword quarter 0/4/8/12
    uint4 vc = *(const uint4*)(xqb + (size_t)(t0 + n) * 16 + q4);
    *(uint4*)&xt[wave][1][n][q4] = vc;
    int tr = t0 - d + n;
    int tc = (tr < 0) ? 0 : tr;
    uint4 vd = *(const uint4*)(xqb + (size_t)tc * 16 + q4);
    if (tr < 0) vd = make_uint4(0, 0, 0, 0);
    *(uint4*)&xt[wave][0][n][q4] = vd;
  }

  v8f acc0, acc1;
#pragma unroll
  for (int r = 0; r < 8; ++r) {
    acc0[r] = bfv[r + 8 * g];
    acc1[r] = bfv[16 + r + 8 * g];
  }
#pragma unroll
  for (int kt = 0; kt < 2; ++kt) {
    v16bf bfr = load_Bfrag_sw(&xt[wave][kt][0][0]);
    acc0 = wmma_bf16(load_Afrag_sw(WfSi + (0 * 2 + kt) * 256), bfr, acc0);
    acc1 = wmma_bf16(load_Afrag_sw(WfSi + (1 * 2 + kt) * 256), bfr, acc1);
  }

  // gate (acc0/acc1 share (m,n) lane layout) -> packed channel pairs
  uint4 up;
  {
    float gv[8];
#pragma unroll
    for (int r = 0; r < 8; ++r)
      gv[r] = fast_tanh(acc0[r]) * fast_sigmoid(acc1[r]);
    up.x = packbf2(gv[0], gv[1]);
    up.y = packbf2(gv[2], gv[3]);
    up.z = packbf2(gv[4], gv[5]);
    up.w = packbf2(gv[6], gv[7]);
  }
  *(uint4*)&gt[wave][n][4 * g] = up;                       // pairs p = 4g..4g+3
  *(uint4*)&gt[wave][n][8 + 4 * g] = make_uint4(0, 0, 0, 0);  // zero K-pad

  // G2: [b][layer][t][p=0..7] dwords -> single b128 store
  *(uint4*)(G2 + (((size_t)(b * NL + layer)) * TT + t0 + n) * 8 + 4 * g) = up;

  // residual: x_out = (Wr @ g + br + x) * invsqrt2 ; emit f32 + packed bf16
  const float* xfi = xf_in + ((size_t)b * RC) * TT + t0;
  float* xfo = xf_out + ((size_t)b * RC) * TT + t0;
  u32* xqo = xq_out + ((size_t)b * TT + t0 + n) * 16;
#pragma unroll
  for (int mt = 0; mt < 2; ++mt) {
    v8f accR;
#pragma unroll
    for (int r = 0; r < 8; ++r) accR[r] = brv[mt * 16 + r + 8 * g];
    accR = wmma_bf16(load_Afrag_sw(WrSi + mt * 256), load_Bfrag_sw(&gt[wave][0][0]), accR);
    v8f xo;
#pragma unroll
    for (int r = 0; r < 8; ++r) {
      int m = mt * 16 + r + 8 * g;
      xo[r] = (accR[r] + xfi[(size_t)m * TT + n]) * 0.70710678118654752f;
      xfo[(size_t)m * TT + n] = xo[r];
    }
    uint4 uq;
    uq.x = packbf2(xo[0], xo[1]);
    uq.y = packbf2(xo[2], xo[3]);
    uq.z = packbf2(xo[4], xo[5]);
    uq.w = packbf2(xo[6], xo[7]);
    *(uint4*)(xqo + mt * 8 + 4 * g) = uq;
  }
}

// ---------------- tail: skip-GEMM + relu + e1 + relu + e2 + transpose -------
__global__ void __launch_bounds__(128) tail_kernel(const u32* __restrict__ G2,
                                                   const u32* __restrict__ WsS,
                                                   const float* __restrict__ bsum,
                                                   const u32* __restrict__ We1S,
                                                   const float* __restrict__ be1,
                                                   const u32* __restrict__ We2S,
                                                   const float* __restrict__ be2,
                                                   float* __restrict__ out) {
  __shared__ __align__(16) u32 Gt[15][16][16];
  __shared__ __align__(16) u32 skT[8][16][16];
  __shared__ __align__(16) u32 h1T[8][16][16];
  __shared__ __align__(16) float outT[16][256];
  const int tid = threadIdx.x;
  const int wave = tid >> 5;
  const int lane = tid & 31;
  const int b = blockIdx.y;
  const int t0 = blockIdx.x * 16;

  // stage G: per (layer, n, half): b128 load -> b128 LDS store
  {
    const u32* Gbase = G2 + ((size_t)b * NL * TT + t0) * 8;
    for (int idx = tid; idx < 960; idx += 128) {
      int i = idx >> 5;                 // layer
      int rem = idx & 31;
      int n2 = rem >> 1;
      int ph = (rem & 1) * 4;           // pair offset within layer row
      uint4 v = *(const uint4*)(Gbase + ((size_t)i * TT + n2) * 8 + ph);
      int pr = i * 8 + ph;              // global pair row
      *(uint4*)&Gt[pr >> 4][n2][pr & 15] = v;
    }
  }
  __syncthreads();

  const int g = (lane >> 4) & 1;
  const int n = lane & 15;

  // skip = relu(Wsall[256x480] @ G + bsum)  (4 Mtiles per wave)
#pragma unroll
  for (int mm = 0; mm < 4; ++mm) {
    const int mt = wave * 4 + mm;
    v8f acc;
#pragma unroll
    for (int r = 0; r < 8; ++r) acc[r] = bsum[mt * 16 + r + 8 * g];
    for (int kt = 0; kt < 15; ++kt)
      acc = wmma_bf16(load_Afrag_sw(WsS + (mt * 15 + kt) * 256),
                      load_Bfrag_sw(&Gt[kt][0][0]), acc);
    uint4 up;
    up.x = packbf2(fmaxf(acc[0], 0.f), fmaxf(acc[1], 0.f));
    up.y = packbf2(fmaxf(acc[2], 0.f), fmaxf(acc[3], 0.f));
    up.z = packbf2(fmaxf(acc[4], 0.f), fmaxf(acc[5], 0.f));
    up.w = packbf2(fmaxf(acc[6], 0.f), fmaxf(acc[7], 0.f));
    *(uint4*)&skT[mt >> 1][n][(mt & 1) * 8 + 4 * g] = up;
  }
  __syncthreads();

  // h1 = relu(We1 @ skip + be1)
#pragma unroll
  for (int mm = 0; mm < 4; ++mm) {
    const int mt = wave * 4 + mm;
    v8f acc;
#pragma unroll
    for (int r = 0; r < 8; ++r) acc[r] = be1[mt * 16 + r + 8 * g];
    for (int kt = 0; kt < 8; ++kt)
      acc = wmma_bf16(load_Afrag_sw(We1S + (mt * 8 + kt) * 256),
                      load_Bfrag_sw(&skT[kt][0][0]), acc);
    uint4 up;
    up.x = packbf2(fmaxf(acc[0], 0.f), fmaxf(acc[1], 0.f));
    up.y = packbf2(fmaxf(acc[2], 0.f), fmaxf(acc[3], 0.f));
    up.z = packbf2(fmaxf(acc[4], 0.f), fmaxf(acc[5], 0.f));
    up.w = packbf2(fmaxf(acc[6], 0.f), fmaxf(acc[7], 0.f));
    *(uint4*)&h1T[mt >> 1][n][(mt & 1) * 8 + 4 * g] = up;
  }
  __syncthreads();

  // logits = We2 @ h1 + be2, staged transposed [t][class]
#pragma unroll
  for (int mm = 0; mm < 4; ++mm) {
    const int mt = wave * 4 + mm;
    v8f acc;
#pragma unroll
    for (int r = 0; r < 8; ++r) acc[r] = be2[mt * 16 + r + 8 * g];
    for (int kt = 0; kt < 8; ++kt)
      acc = wmma_bf16(load_Afrag_sw(We2S + (mt * 8 + kt) * 256),
                      load_Bfrag_sw(&h1T[kt][0][0]), acc);
#pragma unroll
    for (int r = 0; r < 8; ++r) outT[n][mt * 16 + r + 8 * g] = acc[r];
  }
  __syncthreads();

  // coalesced store: out[b][t0+row][0..255]
  for (int idx = tid; idx < 16 * 64; idx += 128) {
    int row = idx >> 6, col4 = idx & 63;
    float4 v = *(const float4*)&outT[row][col4 * 4];
    *(float4*)(out + ((size_t)(b * TT + t0 + row)) * 256 + col4 * 4) = v;
  }
}

// ---------------- host-side launch ------------------------------------------
extern "C" void kernel_launch(void* const* d_in, const int* in_sizes, int n_in,
                              void* d_out, int out_size, void* d_ws, size_t ws_size,
                              hipStream_t stream) {
  (void)in_sizes; (void)n_in; (void)out_size; (void)ws_size;
  const float* y     = (const float*)d_in[0];
  const float* Wc_f  = (const float*)d_in[1];
  const float* bc    = (const float*)d_in[2];
  const float* Wf_f  = (const float*)d_in[3];
  const float* bf_f  = (const float*)d_in[4];
  const float* Wr_f  = (const float*)d_in[5];
  const float* br_f  = (const float*)d_in[6];
  const float* Ws_f  = (const float*)d_in[7];
  const float* bs_f  = (const float*)d_in[8];
  const float* We1_f = (const float*)d_in[9];
  const float* be1   = (const float*)d_in[10];
  const float* We2_f = (const float*)d_in[11];
  const float* be2   = (const float*)d_in[12];
  float* out = (float*)d_out;

  char* ws = (char*)d_ws;
  size_t off = 0;
  auto alloc = [&](size_t bytes) {
    char* p = ws + off;
    off = (off + bytes + 255) & ~(size_t)255;
    return p;
  };
  float* xfA = (float*)alloc(sizeof(float) * BB * RC * TT);
  float* xfB = (float*)alloc(sizeof(float) * BB * RC * TT);
  u32* xqA  = (u32*)alloc(4ull * BB * TT * 16);
  u32* xqB  = (u32*)alloc(4ull * BB * TT * 16);
  u32* G2   = (u32*)alloc(4ull * NL * BB * TT * 8);
  u32* WcS  = (u32*)alloc(4ull * 16 * 256);
  u32* WfS  = (u32*)alloc(4ull * NL * 4 * 256);
  u32* WrS  = (u32*)alloc(4ull * NL * 2 * 256);
  u32* WsS  = (u32*)alloc(4ull * 240 * 256);
  u32* We1S = (u32*)alloc(4ull * 128 * 256);
  u32* We2S = (u32*)alloc(4ull * 128 * 256);
  float* bsum = (float*)alloc(4ull * 256);

  dim3 pg(240, 7);
  prep_kernel<<<pg, 256, 0, stream>>>(Wc_f, Wf_f, Wr_f, Ws_f, We1_f, We2_f, bs_f,
                                      WcS, WfS, WrS, WsS, We1S, We2S, bsum);

  dim3 fg(TT / 32, BB);
  front_kernel<<<fg, 64, 0, stream>>>(y, WcS, bc, xfA, xqA);

  float* xfi = xfA; float* xfo = xfB;
  u32*   xqi = xqA; u32*   xqo = xqB;
  for (int i = 0; i < NL; ++i) {
    int d = 1 << (i % LAYERS);
    dim3 lg(TT / 64, BB);
    layer_kernel<<<lg, 128, 0, stream>>>(xfi, xqi, xfo, xqo, G2,
                                         WfS + i * 4 * 256, bf_f + i * 32,
                                         WrS + i * 2 * 256, br_f + i * 32, i, d);
    float* tf = xfi; xfi = xfo; xfo = tf;
    u32* tq = xqi; xqi = xqo; xqo = tq;
  }

  dim3 tg(TT / 16, BB);
  tail_kernel<<<tg, 128, 0, stream>>>(G2, WsS, bsum, We1S, be1, We2S, be2, out);
}
